// Context2AnswerAttention_1262720385215
// MI455X (gfx1250) — matrix operand order repaired and checked by reading, not compiled, not run on hardware
//
#include <hip/hip_runtime.h>

// ---------------------------------------------------------------------------
// Context2AnswerAttention on MI455X (gfx1250, wave32, WMMA).
//
// B=8, L=2048, N=512, D=H=1024.  ~77 GFLOP vs ~164MB HBM (~7us @ 23.3TB/s)
// -> matrix-pipe bound.  All four einsums run as bf16 WMMA (f32 accum).
// Fused attention core: context_fc / scores / prob live only in LDS.
// Stage-1 weight tiles are staged via CDNA5 async global->LDS copies
// (ASYNCcnt, double-buffered) so each workgroup reads Wt exactly once.
//
// Workspace (bf16, 18 MB): Wt[1024k][1024h], qfcT[8][1024h][512n],
//                          oab[8][512n][1024d]
// ---------------------------------------------------------------------------

typedef __attribute__((ext_vector_type(16))) __bf16        bf16x16;
typedef __attribute__((ext_vector_type(8)))  float         f32x8;
typedef __attribute__((ext_vector_type(4)))  float         f32x4;
typedef __attribute__((ext_vector_type(4)))  unsigned int  u32x4;

#define BATCH 8
#define LCTX  2048
#define NQ    512
#define DIMK  1024      // "dim"
#define HID   1024      // "hidden"
#define TM    64        // context rows per workgroup

#define CLD (HID + 16)  // bf16 elems per cfc row in LDS (2080B rows)
#define SLD (NQ  + 4)   // f32  elems per score row
#define PLD (NQ  + 16)  // bf16 elems per prob row

// LDS layout of the fused kernel (dynamic LDS starts at byte offset 0)
#define CFC_BYTES ((size_t)TM * CLD * 2)          // 133,120
#define S_BYTES   ((size_t)TM * SLD * 4)          // 132,096
#define RED_BYTES ((size_t)TM * 8 * 4)            //   2,048
#define WBUF_OFF  (CFC_BYTES + S_BYTES + RED_BYTES)
#define WROWE 136                                 // padded slab row (bf16 elems)
#define WROWB (WROWE * 2)                         // 272B = 68 dwords (bank-spread)
#define SLAB_B ((size_t)64 * WROWB)               // 17,408 per K=64 slab
#define SMEM_TOTAL (WBUF_OFF + 2 * SLAB_B)        // 302,080 < 320KB/WGP

__device__ __forceinline__ unsigned short f2bf_raw(float f) {
  union { __bf16 b; unsigned short s; } u;
  u.b = (__bf16)f;                 // native v_cvt on gfx1250
  return u.s;
}

union Frag16 {               // wave-striped 16x32 (A) / 32x16 (B) bf16 tile
  bf16x16        v;
  unsigned short s[16];
  u32x4          q[2];
};

__device__ __forceinline__ f32x8 wmma_bf16(bf16x16 a, bf16x16 b, f32x8 c) {
  return __builtin_amdgcn_wmma_f32_16x16x32_bf16(false, a, false, b,
                                                 (short)0, c, false, false);
}

// ---- CDNA5 async global->LDS (per-lane 16B), tracked with ASYNCcnt --------
__device__ __forceinline__ void async_ld16(unsigned lds_byte_off,
                                           const void* gptr) {
  asm volatile("global_load_async_to_lds_b128 %0, %1, off"
               :: "v"(lds_byte_off), "v"((unsigned long long)(size_t)gptr)
               : "memory");
}
__device__ __forceinline__ void wait_async0() {
  asm volatile("s_wait_asynccnt 0x0" ::: "memory");
}
__device__ __forceinline__ void wait_async4() {   // allow next slab in flight
  asm volatile("s_wait_asynccnt 0x4" ::: "memory");
}

// A-fragment (16x32 bf16) from a row-major bf16 LDS tile.
// ISA 7.12.2: lanes 0-15 row M=lane hold K [k0,k0+8)+[k0+16,k0+24);
// lanes 16-31 row M=lane-16 hold K [k0+8,k0+16)+[k0+24,k0+32).
__device__ __forceinline__ bf16x16
load_afrag_lds(const unsigned short* lds, int row0, int ld, int k0, int lane) {
  int half = lane >> 4;
  const unsigned short* p = lds + (row0 + (lane & 15)) * ld + k0 + half * 8;
  Frag16 f;
  f.q[0] = *(const u32x4*)(p);
  f.q[1] = *(const u32x4*)(p + 16);
  return f.v;
}

// Same A-fragment from row-major f32 global, converting on the fly.
__device__ __forceinline__ bf16x16
load_afrag_global_f32(const float* base, int row0, int ld, int k0, int lane) {
  int half = lane >> 4;
  const float* p = base + (long long)(row0 + (lane & 15)) * ld + k0 + half * 8;
  const f32x4* p4 = (const f32x4*)p;
  f32x4 r0 = p4[0], r1 = p4[1];       // K run 1
  f32x4 r2 = p4[4], r3 = p4[5];       // K run 2 (+16 floats)
  Frag16 f;
#pragma unroll
  for (int i = 0; i < 4; ++i) {
    f.s[i]      = f2bf_raw(r0[i]);
    f.s[4 + i]  = f2bf_raw(r1[i]);
    f.s[8 + i]  = f2bf_raw(r2[i]);
    f.s[12 + i] = f2bf_raw(r3[i]);
  }
  return f.v;
}

// B-fragment (32x16 bf16): lane <-> N col (col0+lane%16); K [k0+8h,+8)+[k0+16+8h,+8).
__device__ __forceinline__ bf16x16
load_bfrag_global(const unsigned short* base, int k0, int ld, int col0, int lane) {
  const unsigned short* p = base + (long long)(k0 + (lane >> 4) * 8) * ld + col0 + (lane & 15);
  Frag16 f;
#pragma unroll
  for (int i = 0; i < 8; ++i) {
    f.s[i]     = p[(long long)i * ld];
    f.s[8 + i] = p[(long long)(16 + i) * ld];
  }
  return f.v;
}

__device__ __forceinline__ bf16x16
load_bfrag_lds(const unsigned short* lds, int k0, int ld, int col0, int lane) {
  const unsigned short* p = lds + (k0 + (lane >> 4) * 8) * ld + col0 + (lane & 15);
  Frag16 f;
#pragma unroll
  for (int i = 0; i < 8; ++i) {
    f.s[i]     = p[i * ld];
    f.s[8 + i] = p[(16 + i) * ld];
  }
  return f.v;
}

// ---------------------------------------------------------------------------
// K0a: W (f32 [h][k]) -> Wt (bf16 [k][h]), LDS-tiled transpose (coalesced).
__global__ __launch_bounds__(256) void conv_wt(const float* __restrict__ W,
                                               unsigned short* __restrict__ Wt) {
  __shared__ float t[32][33];
  const int bx = blockIdx.x, by = blockIdx.y;       // h-tile, k-tile
  const int x = threadIdx.x & 31, y = threadIdx.x >> 5;
#pragma unroll
  for (int dy = 0; dy < 32; dy += 8)                // read W[h][k], k coalesced
    t[y + dy][x] = W[(bx * 32 + y + dy) * DIMK + by * 32 + x];
  __syncthreads();
#pragma unroll
  for (int dy = 0; dy < 32; dy += 8)                // write Wt[k][h], h coalesced
    Wt[(by * 32 + y + dy) * HID + bx * 32 + x] = f2bf_raw(t[x][y + dy]);
}

// K0b: out_answers f32 -> bf16
__global__ __launch_bounds__(256) void conv_oa(const float* __restrict__ src,
                                               unsigned short* __restrict__ dst,
                                               int n) {
  int idx = blockIdx.x * 256 + threadIdx.x;
  if (idx < n) dst[idx] = f2bf_raw(src[idx]);
}

// ---------------------------------------------------------------------------
// K1: qfcT[b][h][n] = relu(answers[b][n] . W[h])  (bf16, transposed store)
__global__ __launch_bounds__(256) void qfc_kernel(
    const float* __restrict__ answers,        // (B, 512, 1024)
    const unsigned short* __restrict__ Wt,    // (1024 k, 1024 h)
    unsigned short* __restrict__ qfcT) {      // (B, 1024 h, 512 n)
  const int b  = blockIdx.z;
  const int n0 = blockIdx.x * 64;
  const int h0 = blockIdx.y * 128;
  const int wave = threadIdx.x >> 5, lane = threadIdx.x & 31;
  const int mi = wave & 3, wj = wave >> 2;
  const int hbase = h0 + wj * 64;

  const float* ansB = answers + ((long long)b * NQ + n0) * DIMK;

  f32x8 acc[4];
#pragma unroll
  for (int i = 0; i < 4; ++i)
#pragma unroll
    for (int r = 0; r < 8; ++r) acc[i][r] = 0.0f;

  for (int k0 = 0; k0 < DIMK; k0 += 32) {
    bf16x16 a = load_afrag_global_f32(ansB, mi * 16, DIMK, k0, lane);
#pragma unroll
    for (int nc = 0; nc < 4; ++nc) {
      bf16x16 bm = load_bfrag_global(Wt, k0, HID, hbase + nc * 16, lane);
      acc[nc] = wmma_bf16(a, bm, acc[nc]);
    }
  }

  unsigned short* qB = qfcT + (long long)b * HID * NQ;
  const int M0 = mi * 16 + (lane >> 4) * 8;
  const int N  = lane & 15;
#pragma unroll
  for (int nc = 0; nc < 4; ++nc) {
    int h = hbase + nc * 16 + N;
#pragma unroll
    for (int r = 0; r < 8; ++r) {
      float v = acc[nc][r];
      v = v > 0.0f ? v : 0.0f;
      qB[(long long)h * NQ + (n0 + M0 + r)] = f2bf_raw(v);
    }
  }
}

// ---------------------------------------------------------------------------
// K2: fused relu(ctx@W^T) -> scores -> mask -> softmax -> @out_answers.
__global__ __launch_bounds__(256) void attn_fused(
    const float* __restrict__ ctx,            // (B, L, 1024)
    const int* __restrict__ ans_mask,         // (B, 512)
    const unsigned short* __restrict__ Wt,    // (1024, 1024) bf16
    const unsigned short* __restrict__ qfcT,  // (B, 1024, 512) bf16
    const unsigned short* __restrict__ oab,   // (B, 512, 1024) bf16
    float* __restrict__ out) {                // (B, L, 1024)
  extern __shared__ char smem[];
  unsigned short* cfc = (unsigned short*)smem;             // [64][CLD] bf16
  unsigned short* Pb  = (unsigned short*)smem;             // alias: [64][PLD]
  float* S   = (float*)(smem + CFC_BYTES);                 // [64][SLD]
  float* red = (float*)(smem + CFC_BYTES + S_BYTES);       // [64][8]

  const int b  = blockIdx.y;
  const long long l0 = (long long)blockIdx.x * TM;
  const int tid = threadIdx.x;
  const int wave = tid >> 5, lane = tid & 31;
  const int mi = wave & 3;   // 16-row M tile
  const int wj = wave >> 2;  // 0..1

  const float* ctxB = ctx + ((long long)b * LCTX + l0) * DIMK;

  // ---- Stage 1: cfc = relu(ctx_tile @ W^T), Wt slabs async-staged ---------
  for (int h0 = 0; h0 < HID; h0 += 128) {
    f32x8 acc[4];
#pragma unroll
    for (int i = 0; i < 4; ++i)
#pragma unroll
      for (int r = 0; r < 8; ++r) acc[i][r] = 0.0f;

    // prologue: stage slab 0 (K rows 0..63 of the 128 H columns at h0)
#pragma unroll
    for (int it = 0; it < 4; ++it) {
      int chunk = it * 256 + tid;                 // 1024 x 16B chunks
      int kk = chunk >> 4, cc = chunk & 15;
      async_ld16((unsigned)(WBUF_OFF + kk * WROWB + cc * 16),
                 Wt + (size_t)kk * HID + h0 + cc * 8);
    }

    for (int s = 0; s < DIMK / 64; ++s) {         // 16 K-slabs of 64
      const int buf = s & 1;
      if (s + 1 < DIMK / 64) {                    // stage next slab
        const int nbuf = (s + 1) & 1;
#pragma unroll
        for (int it = 0; it < 4; ++it) {
          int chunk = it * 256 + tid;
          int kk = chunk >> 4, cc = chunk & 15;
          async_ld16((unsigned)(WBUF_OFF + nbuf * SLAB_B + kk * WROWB + cc * 16),
                     Wt + (size_t)((s + 1) * 64 + kk) * HID + h0 + cc * 8);
        }
        wait_async4();                            // slab s done, s+1 in flight
      } else {
        wait_async0();
      }
      __syncthreads();                            // slab s visible to all waves

      const unsigned short* wbuf =
          (const unsigned short*)(smem + WBUF_OFF + buf * SLAB_B);
#pragma unroll
      for (int kk = 0; kk < 64; kk += 32) {
        bf16x16 a = load_afrag_global_f32(ctxB, mi * 16, DIMK, s * 64 + kk, lane);
#pragma unroll
        for (int nc = 0; nc < 4; ++nc) {
          bf16x16 bm = load_bfrag_lds(wbuf, kk, WROWE, wj * 64 + nc * 16, lane);
          acc[nc] = wmma_bf16(a, bm, acc[nc]);
        }
      }
      __syncthreads();                            // done reading buf before reuse
    }

    const int hbase = h0 + wj * 64;
    const int M0 = mi * 16 + (lane >> 4) * 8;
    const int N  = lane & 15;
#pragma unroll
    for (int nc = 0; nc < 4; ++nc)
#pragma unroll
      for (int r = 0; r < 8; ++r) {
        float v = acc[nc][r];
        v = v > 0.0f ? v : 0.0f;
        cfc[(M0 + r) * CLD + hbase + nc * 16 + N] = f2bf_raw(v);
      }
  }
  __syncthreads();

  // ---- Stage 2: S = cfc @ qfcT (+mask) -> LDS f32 -------------------------
  const unsigned short* qT = qfcT + (long long)b * HID * NQ;
  for (int n0 = 0; n0 < NQ; n0 += 64) {
    f32x8 acc[2];
#pragma unroll
    for (int i = 0; i < 2; ++i)
#pragma unroll
      for (int r = 0; r < 8; ++r) acc[i][r] = 0.0f;

    for (int k0 = 0; k0 < HID; k0 += 32) {
      bf16x16 a = load_afrag_lds(cfc, mi * 16, CLD, k0, lane);
#pragma unroll
      for (int nc = 0; nc < 2; ++nc) {
        bf16x16 bm = load_bfrag_global(qT, k0, NQ, n0 + wj * 32 + nc * 16, lane);
        acc[nc] = wmma_bf16(a, bm, acc[nc]);
      }
    }
    const int M0 = mi * 16 + (lane >> 4) * 8;
    const int N  = lane & 15;
#pragma unroll
    for (int nc = 0; nc < 2; ++nc) {
      int col = n0 + wj * 32 + nc * 16 + N;
      int alive = ans_mask[b * NQ + col];
#pragma unroll
      for (int r = 0; r < 8; ++r)
        S[(M0 + r) * SLD + col] = alive ? acc[nc][r] : -1e9f;
    }
  }
  __syncthreads();

  // ---- Stage 3: exact softmax over N=512; prob bf16 aliases cfc -----------
  {
    const int row = tid & 63, q = tid >> 6;       // 4 threads per row
    float* srow = S + row * SLD + q * 128;
    float mx = -3.0e38f;
    for (int i = 0; i < 128; ++i) mx = fmaxf(mx, srow[i]);
    red[row * 8 + q] = mx;
    __syncthreads();
    mx = fmaxf(fmaxf(red[row * 8 + 0], red[row * 8 + 1]),
               fmaxf(red[row * 8 + 2], red[row * 8 + 3]));
    float ps = 0.0f;
    for (int i = 0; i < 128; ++i) {
      float e = __expf(srow[i] - mx);
      srow[i] = e;
      ps += e;
    }
    red[row * 8 + 4 + q] = ps;
    __syncthreads();
    float inv = 1.0f / (red[row * 8 + 4] + red[row * 8 + 5] +
                        red[row * 8 + 6] + red[row * 8 + 7]);
    unsigned short* prow = Pb + row * PLD + q * 128;
    for (int i = 0; i < 128; ++i) prow[i] = f2bf_raw(srow[i] * inv);
  }
  __syncthreads();

  // ---- Stage 4: out = P @ out_answers -------------------------------------
  const unsigned short* oaB = oab + (long long)b * NQ * DIMK;
  float* outB = out + ((long long)b * LCTX + l0) * DIMK;
  for (int d0 = 0; d0 < DIMK; d0 += 128) {
    const int dbase = d0 + wj * 64;
    f32x8 acc[4];
#pragma unroll
    for (int i = 0; i < 4; ++i)
#pragma unroll
      for (int r = 0; r < 8; ++r) acc[i][r] = 0.0f;

    for (int k0 = 0; k0 < NQ; k0 += 32) {
      bf16x16 a = load_afrag_lds(Pb, mi * 16, PLD, k0, lane);
#pragma unroll
      for (int nc = 0; nc < 4; ++nc) {
        bf16x16 bm = load_bfrag_global(oaB, k0, DIMK, dbase + nc * 16, lane);
        acc[nc] = wmma_bf16(a, bm, acc[nc]);
      }
    }
    const int M0 = mi * 16 + (lane >> 4) * 8;
    const int N  = lane & 15;
#pragma unroll
    for (int nc = 0; nc < 4; ++nc)
#pragma unroll
      for (int r = 0; r < 8; ++r)
        outB[(long long)(M0 + r) * DIMK + dbase + nc * 16 + N] = acc[nc][r];
  }
}

// ---------------------------------------------------------------------------
extern "C" void kernel_launch(void* const* d_in, const int* in_sizes, int n_in,
                              void* d_out, int out_size, void* d_ws, size_t ws_size,
                              hipStream_t stream) {
  (void)in_sizes; (void)n_in; (void)out_size; (void)ws_size;
  const float* ctx         = (const float*)d_in[0];  // (8,2048,1024)
  const float* answers     = (const float*)d_in[1];  // (8,512,1024)
  const float* out_answers = (const float*)d_in[2];  // (8,512,1024)
  const int*   ans_mask    = (const int*)d_in[3];    // (8,512)
  const float* W           = (const float*)d_in[4];  // (1024,1024)
  float* out = (float*)d_out;                        // (8,2048,1024)

  unsigned short* Wt   = (unsigned short*)d_ws;                 // 1024*1024
  unsigned short* qfcT = Wt + (size_t)HID * DIMK;               // 8*1024*512
  unsigned short* oab  = qfcT + (size_t)BATCH * HID * NQ;       // 8*512*1024

  dim3 gt(HID / 32, DIMK / 32);
  conv_wt<<<gt, 256, 0, stream>>>(W, Wt);
  conv_oa<<<(BATCH * NQ * DIMK) / 256, 256, 0, stream>>>(
      out_answers, oab, BATCH * NQ * DIMK);

  dim3 g1(NQ / 64, HID / 128, BATCH);
  qfc_kernel<<<g1, 256, 0, stream>>>(answers, Wt, qfcT);

  dim3 g2(LCTX / TM, BATCH);
  attn_fused<<<g2, 256, SMEM_TOTAL, stream>>>(ctx, ans_mask, Wt, qfcT, oab, out);
}
